// GCNNetwork_32478542693014
// MI455X (gfx1250) — compile-verified, matrix operand order checked by reading, not run on hardware
//
#include <hip/hip_runtime.h>
#include <hip/hip_bf16.h>
#include <math.h>

#define N_NODES 60000
#define N_EDGES 120000
#define NBATCH  512
#define DD      256
#define LLAY    6
#define ZDIM    (LLAY * DD)   // 1536

typedef __attribute__((ext_vector_type(16))) __bf16 v16bf;
typedef __attribute__((ext_vector_type(8)))  __bf16 v8bf;
typedef __attribute__((ext_vector_type(8)))  float  v8f;

union BF16x16 { v16bf v; v8bf h[2]; };

__device__ __forceinline__ float wave_sum32(float v) {
  #pragma unroll
  for (int o = 16; o > 0; o >>= 1) v += __shfl_xor(v, o, 32);
  return v;
}

__device__ __forceinline__ float gelu_exact(float x) {
  return 0.5f * x * (1.0f + erff(x * 0.70710678118654752f));
}

// ---------------- element-wise / scatter kernels ----------------

__global__ void zero_f32_kernel(float* p, size_t n) {
  size_t i = (size_t)blockIdx.x * blockDim.x + threadIdx.x;
  size_t stride = (size_t)gridDim.x * blockDim.x;
  for (; i < n; i += stride) p[i] = 0.0f;
}

__global__ void embed_kernel(const int* __restrict__ x,
                             const float* __restrict__ emb,
                             float* __restrict__ h) {
  size_t i = (size_t)blockIdx.x * 256 + threadIdx.x;   // N*D threads
  int n = (int)(i >> 8), d = (int)(i & 255);
  h[i] = emb[(size_t)x[n] * DD + d];
}

// one wave per node, 8 elements per lane
__global__ __launch_bounds__(256) void ln_kernel(const float* __restrict__ h,
                                                 const float* __restrict__ scale,
                                                 const float* __restrict__ bias,
                                                 float* __restrict__ y) {
  int wave = threadIdx.x >> 5, lane = threadIdx.x & 31;
  int n = blockIdx.x * 8 + wave;
  if (n >= N_NODES) return;
  const float* hr = h + (size_t)n * DD;
  float v[8]; float s = 0.0f;
  #pragma unroll
  for (int j = 0; j < 8; ++j) { v[j] = hr[lane + 32 * j]; s += v[j]; }
  s = wave_sum32(s);
  float mu = s * (1.0f / DD);
  float q = 0.0f;
  #pragma unroll
  for (int j = 0; j < 8; ++j) { float dlt = v[j] - mu; q += dlt * dlt; }
  q = wave_sum32(q);
  float rs = rsqrtf(q * (1.0f / DD) + 1e-5f);
  float* yr = y + (size_t)n * DD;
  #pragma unroll
  for (int j = 0; j < 8; ++j) {
    int d = lane + 32 * j;
    yr[d] = (v[j] - mu) * rs * scale[d] + bias[d];
  }
}

// msg >= eps > 0 so uint-punned max == float max
__global__ void edge_max_kernel(const int* __restrict__ ei,
                                const float* __restrict__ ea,
                                const float* __restrict__ y,
                                const float* __restrict__ wlw,
                                const float* __restrict__ wlb,
                                unsigned int* __restrict__ mmax) {
  size_t i = (size_t)blockIdx.x * 256 + threadIdx.x;   // E*D threads
  int e = (int)(i >> 8), d = (int)(i & 255);
  int s = ei[e];
  int t = ei[N_EDGES + e];
  float msg = fmaxf(y[(size_t)s * DD + d] + ea[e] * wlw[d] + wlb[d], 0.0f) + 1e-7f;
  atomicMax(&mmax[(size_t)t * DD + d], __float_as_uint(msg));
}

__global__ void edge_sum_kernel(const int* __restrict__ ei,
                                const float* __restrict__ ea,
                                const float* __restrict__ y,
                                const float* __restrict__ wlw,
                                const float* __restrict__ wlb,
                                const float* __restrict__ mmax,
                                float* __restrict__ den,
                                float* __restrict__ num) {
  size_t i = (size_t)blockIdx.x * 256 + threadIdx.x;
  int e = (int)(i >> 8), d = (int)(i & 255);
  int s = ei[e];
  int t = ei[N_EDGES + e];
  float msg = fmaxf(y[(size_t)s * DD + d] + ea[e] * wlw[d] + wlb[d], 0.0f) + 1e-7f;
  size_t ti = (size_t)t * DD + d;
  float w = expf(msg - mmax[ti]);
  atomicAdd(&den[ti], w);
  atomicAdd(&num[ti], msg * w);
}

__global__ void combine_kernel(const float* __restrict__ num,
                               const float* __restrict__ den,
                               const float* __restrict__ y,
                               __bf16* __restrict__ tb) {
  size_t i = (size_t)blockIdx.x * 256 + threadIdx.x;   // N*D threads
  float agg = num[i] / fmaxf(den[i], 1e-16f);
  tb[i] = (__bf16)(agg + y[i]);
}

__global__ void pool_kernel(const float* __restrict__ h,
                            const int* __restrict__ batch,
                            float* __restrict__ z, int lofs) {
  size_t i = (size_t)blockIdx.x * 256 + threadIdx.x;   // N*D threads
  int n = (int)(i >> 8), d = (int)(i & 255);
  atomicAdd(&z[(size_t)batch[n] * ZDIM + lofs + d], h[i]);
}

// w [K][Ncol] f32 row-major -> wt [Ncol][K] bf16
__global__ void transpose_cvt_kernel(const float* __restrict__ w,
                                     __bf16* __restrict__ wt, int K, int Ncol) {
  size_t i = (size_t)blockIdx.x * 256 + threadIdx.x;
  if (i >= (size_t)K * Ncol) return;
  int k = (int)(i / Ncol), n = (int)(i % Ncol);
  wt[(size_t)n * K + k] = (__bf16)w[i];
}

__global__ void cvt_bf16_kernel(const float* __restrict__ a,
                                __bf16* __restrict__ b, size_t n) {
  size_t i = (size_t)blockIdx.x * 256 + threadIdx.x;
  if (i < n) b[i] = (__bf16)a[i];
}

// ---------------- WMMA GEMM ----------------
// C[M x Ncol] = act(A[M x K] @ Bt[Ncol x K]^T + bias)
// A, Bt bf16 row-major with ld = K. Each wave: 16x64 tile.
// ACT: 0 = relu, 1 = gelu(exact).  OUTB: 0 -> f32 C, 1 -> bf16 C.
template <int ACT, int OUTB>
__global__ __launch_bounds__(256) void wmma_gemm_kernel(
    const __bf16* __restrict__ A, const __bf16* __restrict__ Bt,
    const float* __restrict__ bias, float* __restrict__ Cf,
    __bf16* __restrict__ Cb, int M, int K, int Ncol) {
  const int lane = threadIdx.x & 31;
  const int wave = threadIdx.x >> 5;
  const int r = lane & 15;
  const int hi = lane >> 4;
  const int row0 = (blockIdx.y * 8 + wave) * 16;
  const int col0 = blockIdx.x * 64;
  if (row0 >= M) return;

  v8f acc[4] = {v8f{}, v8f{}, v8f{}, v8f{}};
  const __bf16* arow = A + (size_t)(row0 + r) * K + 8 * hi;
  const __bf16* bp[4];
  #pragma unroll
  for (int t = 0; t < 4; ++t)
    bp[t] = Bt + (size_t)(col0 + 16 * t + r) * K + 8 * hi;

  for (int k0 = 0; k0 < K; k0 += 32) {
    BF16x16 a;
    a.h[0] = *(const v8bf*)(arow + k0);
    a.h[1] = *(const v8bf*)(arow + k0 + 16);
    #pragma unroll
    for (int t = 0; t < 4; ++t) {
      BF16x16 b;
      b.h[0] = *(const v8bf*)(bp[t] + k0);
      b.h[1] = *(const v8bf*)(bp[t] + k0 + 16);
      acc[t] = __builtin_amdgcn_wmma_f32_16x16x32_bf16(
          false, a.v, false, b.v, (short)0, acc[t], false, false);
    }
  }

  #pragma unroll
  for (int t = 0; t < 4; ++t) {
    int c = col0 + 16 * t + r;
    float bs = bias[c];
    #pragma unroll
    for (int j = 0; j < 8; ++j) {
      float v = acc[t][j] + bs;
      if (ACT == 0) v = fmaxf(v, 0.0f);
      else          v = gelu_exact(v);
      size_t idx = (size_t)(row0 + j + 8 * hi) * Ncol + c;
      if (OUTB) Cb[idx] = (__bf16)v;
      else      Cf[idx] = v;
    }
  }
}

// out[b] = dot(z3[b, 0:192], w3) + b3 ; one wave per row
__global__ __launch_bounds__(256) void final_kernel(const __bf16* __restrict__ z3,
                                                    const float* __restrict__ w3,
                                                    const float* __restrict__ b3,
                                                    float* __restrict__ out) {
  int wave = threadIdx.x >> 5, lane = threadIdx.x & 31;
  int b = blockIdx.x * 8 + wave;
  if (b >= NBATCH) return;
  float s = 0.0f;
  #pragma unroll
  for (int j = 0; j < 6; ++j) {
    int d = lane + 32 * j;
    s += (float)z3[(size_t)b * 192 + d] * w3[d];
  }
  s = wave_sum32(s);
  if (lane == 0) out[b] = s + b3[0];
}

// ---------------- host launch ----------------

extern "C" void kernel_launch(void* const* d_in, const int* in_sizes, int n_in,
                              void* d_out, int out_size, void* d_ws, size_t ws_size,
                              hipStream_t stream) {
  const int*   x        = (const int*)  d_in[0];
  const int*   ei       = (const int*)  d_in[1];
  const float* ea       = (const float*)d_in[2];
  const int*   batch    = (const int*)  d_in[3];
  const float* node_emb = (const float*)d_in[4];
  const float* ln_s     = (const float*)d_in[5];
  const float* ln_b     = (const float*)d_in[6];
  const float* wl_w     = (const float*)d_in[7];
  const float* wl_b     = (const float*)d_in[8];
  const float* conv_w   = (const float*)d_in[9];
  const float* conv_b   = (const float*)d_in[10];
  const float* ro_w0 = (const float*)d_in[11]; const float* ro_b0 = (const float*)d_in[12];
  const float* ro_w1 = (const float*)d_in[13]; const float* ro_b1 = (const float*)d_in[14];
  const float* ro_w2 = (const float*)d_in[15]; const float* ro_b2 = (const float*)d_in[16];
  const float* ro_w3 = (const float*)d_in[17]; const float* ro_b3 = (const float*)d_in[18];
  float* out = (float*)d_out;

  char* ws = (char*)d_ws;
  size_t off = 0;
  auto take = [&](size_t bytes) { char* p = ws + off; off = (off + bytes + 255) & ~(size_t)255; return p; };
  const size_t ND = (size_t)N_NODES * DD;

  float*        h    = (float*)       take(ND * 4);
  float*        y    = (float*)       take(ND * 4);
  float*        mnum = (float*)       take(ND * 4 * 3);  // [mmax | den | num] contiguous
  unsigned int* mmax = (unsigned int*)mnum;
  float*        den  = mnum + ND;
  float*        num  = mnum + 2 * ND;
  __bf16*       tb   = (__bf16*)      take(ND * 2);
  float*        z    = (float*)       take((size_t)NBATCH * ZDIM * 4);
  __bf16*       zb   = (__bf16*)      take((size_t)NBATCH * ZDIM * 2);
  __bf16*       z1   = (__bf16*)      take((size_t)NBATCH * 768 * 2);
  __bf16*       z2   = (__bf16*)      take((size_t)NBATCH * 384 * 2);
  __bf16*       z3   = (__bf16*)      take((size_t)NBATCH * 192 * 2);
  __bf16*       wct  = (__bf16*)      take((size_t)LLAY * DD * DD * 2);
  __bf16*       w0t  = (__bf16*)      take((size_t)1536 * 768 * 2);
  __bf16*       w1t  = (__bf16*)      take((size_t)768 * 384 * 2);
  __bf16*       w2t  = (__bf16*)      take((size_t)384 * 192 * 2);
  (void)ws_size; (void)in_sizes; (void)n_in; (void)out_size;

  // weight preprocessing: transpose + convert to bf16
  for (int l = 0; l < LLAY; ++l)
    transpose_cvt_kernel<<<(DD * DD + 255) / 256, 256, 0, stream>>>(
        conv_w + (size_t)l * DD * DD, wct + (size_t)l * DD * DD, DD, DD);
  transpose_cvt_kernel<<<((size_t)1536 * 768 + 255) / 256, 256, 0, stream>>>(ro_w0, w0t, 1536, 768);
  transpose_cvt_kernel<<<((size_t)768 * 384 + 255) / 256, 256, 0, stream>>>(ro_w1, w1t, 768, 384);
  transpose_cvt_kernel<<<((size_t)384 * 192 + 255) / 256, 256, 0, stream>>>(ro_w2, w2t, 384, 192);

  zero_f32_kernel<<<2048, 256, 0, stream>>>(z, (size_t)NBATCH * ZDIM);
  embed_kernel<<<N_NODES, 256, 0, stream>>>(x, node_emb, h);

  for (int l = 0; l < LLAY; ++l) {
    const float* ws_ = ln_s + (size_t)l * DD;
    const float* wb_ = ln_b + (size_t)l * DD;
    const float* ww  = wl_w + (size_t)l * DD;
    const float* wbb = wl_b + (size_t)l * DD;
    const float* cb  = conv_b + (size_t)l * DD;
    const __bf16* cw = wct + (size_t)l * DD * DD;

    zero_f32_kernel<<<4096, 256, 0, stream>>>(mnum, ND * 3);
    ln_kernel<<<N_NODES / 8, 256, 0, stream>>>(h, ws_, wb_, y);
    edge_max_kernel<<<N_EDGES, 256, 0, stream>>>(ei, ea, y, ww, wbb, mmax);
    edge_sum_kernel<<<N_EDGES, 256, 0, stream>>>(ei, ea, y, ww, wbb, (const float*)mnum, den, num);
    combine_kernel<<<N_NODES, 256, 0, stream>>>(num, den, y, tb);
    {
      dim3 grid(DD / 64, (N_NODES + 127) / 128);
      wmma_gemm_kernel<0, 0><<<grid, 256, 0, stream>>>(tb, cw, cb, h, (
          __bf16*)nullptr, N_NODES, DD, DD);
    }
    pool_kernel<<<N_NODES, 256, 0, stream>>>(h, batch, z, l * DD);
  }

  // readout
  cvt_bf16_kernel<<<((size_t)NBATCH * ZDIM + 255) / 256, 256, 0, stream>>>(z, zb, (size_t)NBATCH * ZDIM);
  {
    dim3 g0(768 / 64, NBATCH / 128);
    wmma_gemm_kernel<1, 1><<<g0, 256, 0, stream>>>(zb, w0t, ro_b0, (float*)nullptr, z1, NBATCH, ZDIM, 768);
    dim3 g1(384 / 64, NBATCH / 128);
    wmma_gemm_kernel<1, 1><<<g1, 256, 0, stream>>>(z1, w1t, ro_b1, (float*)nullptr, z2, NBATCH, 768, 384);
    dim3 g2(192 / 64, NBATCH / 128);
    wmma_gemm_kernel<1, 1><<<g2, 256, 0, stream>>>(z2, w2t, ro_b2, (float*)nullptr, z3, NBATCH, 384, 192);
  }
  final_kernel<<<NBATCH / 8, 256, 0, stream>>>(z3, ro_w3, ro_b3, out);
}
